// EdgePredictor_63574105915435
// MI455X (gfx1250) — compile-verified, matrix operand order checked by reading, not run on hardware
//
#include <hip/hip_runtime.h>

typedef _Float16 f16;
typedef __attribute__((ext_vector_type(16))) _Float16 v16h;
typedef __attribute__((ext_vector_type(8)))  _Float16 v8h;
typedef __attribute__((ext_vector_type(8)))  float    v8f;
typedef __attribute__((ext_vector_type(4)))  unsigned v4u;
typedef __attribute__((ext_vector_type(8)))  int      v8i;
typedef __attribute__((ext_vector_type(4)))  int      v4i;

// Shapes (fixed by the reference)
// latent: 8x256x256, positions: 8x256x3, atom: 8x256x16
// W1,W2: 256x256, W3: 273x256, W4: 256x4
// out: 8x256x256x4 f32

#if __has_builtin(__builtin_amdgcn_tensor_load_to_lds) && __has_builtin(__builtin_amdgcn_s_wait_tensorcnt)
#define USE_TDM 1
#endif

static __device__ __forceinline__ v16h pack16(v8h a, v8h b) {
    return __builtin_shufflevector(a, b, 0,1,2,3,4,5,6,7,8,9,10,11,12,13,14,15);
}
static __device__ __forceinline__ v8f wmma_f16(v16h a, v16h b, v8f c) {
    return __builtin_amdgcn_wmma_f32_16x16x32_f16(false, a, false, b, (short)0, c, false, false);
}

#ifdef USE_TDM
// TDM DMA of one swizzled-W3 N-quarter into LDS:
// 2D tile, 9 rows (K-chunks) x 4096B, global row stride 16384B, packed
// contiguously into LDS (36864B). data_size = 8B units.
static __device__ __forceinline__ void tdm_load_quarter(const f16* gsrc, f16* lds_dst) {
    unsigned long long ga = (unsigned long long)gsrc;
    unsigned lds = (unsigned)(unsigned long long)lds_dst;   // LDS aperture low bits = byte offset
    v4u g0;
    g0.x = 1u;                                              // count=1 (valid), user mode
    g0.y = lds;                                             // lds_addr [63:32]
    g0.z = (unsigned)(ga & 0xFFFFFFFFu);                    // global_addr [95:64]
    g0.w = (unsigned)((ga >> 32) & 0x01FFFFFFu) | (2u << 30); // addr[56:32] | type=2
    v8i g1;
    g1[0] = (int)(3u << 16);       // wg_mask=0, data_size=3 (8B), no iterate/pad/barrier
    g1[1] = (int)(2048u << 16);    // tensor_dim0[15:0]=2048 (8B units) in bits[63:48]
    g1[2] = (int)(16u << 16);      // tensor_dim0 hi=0 | tensor_dim1=16 in bits[95:80]
    g1[3] = (int)(512u << 16);     // tensor_dim1 hi=0 | tile_dim0=512 in bits[127:112]
    g1[4] = 9;                     // tile_dim1=9, tile_dim2=0
    g1[5] = 2048;                  // tensor_dim0_stride[31:0] = 2048 (8B units = 16KB)
    g1[6] = 0;                     // stride0 hi | tensor_dim1_stride lo
    g1[7] = 0;                     // tensor_dim1_stride hi
    v4i gz = {0, 0, 0, 0};         // groups 2/3: zero (<=2D tensor, dims unused)
#if __clang_major__ >= 23
    v8i z8 = {0, 0, 0, 0, 0, 0, 0, 0};
    __builtin_amdgcn_tensor_load_to_lds(g0, g1, gz, gz, z8, 0);
#else
    __builtin_amdgcn_tensor_load_to_lds(g0, g1, gz, gz, 0);
#endif
}
#endif

// ---------------------------------------------------------------------------
// Setup: convert latent + atom_types to f16
__global__ void k_cvt(const float* __restrict__ lat, const float* __restrict__ at,
                      f16* __restrict__ lat16, f16* __restrict__ at16) {
    int idx = blockIdx.x * 256 + threadIdx.x;
    if (idx < 524288) {
        lat16[idx] = (f16)lat[idx];
    } else {
        int a = idx - 524288;
        if (a < 32768) at16[a] = (f16)at[a];
    }
}

// Swizzle W1/W2 (256x256 f32, row=K col=N) into 16-bit WMMA B-fragment layout:
// dst[((kc*16+nt)*32+lane)*16+t], K_local = (lane<16?0:8) + (t<8?t:t+8)
__global__ void k_swzW12(const float* __restrict__ W1, const float* __restrict__ W2,
                         f16* __restrict__ W1s, f16* __restrict__ W2s) {
    int idx = blockIdx.x * 256 + threadIdx.x;   // 0..131071
    int e    = idx & 65535;
    int t    = e & 15;
    int lane = (e >> 4) & 31;
    int nt   = (e >> 9) & 15;
    int kc   = e >> 13;                         // 0..7
    int kl   = ((lane < 16) ? 0 : 8) + ((t < 8) ? t : (t + 8));
    int row  = kc * 32 + kl;
    int col  = nt * 16 + (lane & 15);
    float v  = (idx < 65536) ? W1[row * 256 + col] : W2[row * 256 + col];
    if (idx < 65536) W1s[e] = (f16)v; else W2s[e] = (f16)v;
}

// Swizzle W3 (273x256) into B-fragment layout, K padded to 288:
// chunk 0: K=0 -> W3 row 0 (dist), K=1..16 -> rows 1..16 (atom), K=17..31 -> 0
// chunks 1..8: K_local -> W3 row 17 + (kc-1)*32 + K_local (latent part)
__global__ void k_swzW3(const float* __restrict__ W3, f16* __restrict__ W3s) {
    int idx = blockIdx.x * 256 + threadIdx.x;   // 0..73727 (9*16*32*16)
    int t    = idx & 15;
    int lane = (idx >> 4) & 31;
    int nt   = (idx >> 9) & 15;
    int kc   = idx >> 13;                       // 0..8
    int kl   = ((lane < 16) ? 0 : 8) + ((t < 8) ? t : (t + 8));
    int col  = nt * 16 + (lane & 15);
    float v;
    if (kc == 0) {
        if (kl == 0)       v = W3[col];
        else if (kl <= 16) v = W3[kl * 256 + col];
        else               v = 0.f;
    } else {
        int row = 17 + (kc - 1) * 32 + kl;
        v = W3[row * 256 + col];
    }
    W3s[idx] = (f16)v;
}

// ---------------------------------------------------------------------------
// Latent MLP layer: out16 = (relu?)(A16 @ W + bias), M=2048, K=256, N=256.
// One 16x16 tile per wave, 8 waves/block, 256 blocks.
__global__ void k_mlp(const f16* __restrict__ A, const f16* __restrict__ Bs,
                      const float* __restrict__ bias, f16* __restrict__ out,
                      int doRelu) {
    int tid = threadIdx.x;
    int L = tid & 31, wv = tid >> 5;
    int Lm = L & 15, lh = L >> 4;
    int tileId = blockIdx.x * 8 + wv;           // 0..2047
    int mt = tileId >> 4, nt = tileId & 15;

    const f16* arow = A + (mt * 16 + Lm) * 256;
    v8f acc = {};
#pragma unroll
    for (int kc = 0; kc < 8; ++kc) {
        int c0 = kc * 32 + lh * 8;
        v16h af = pack16(*(const v8h*)(arow + c0), *(const v8h*)(arow + c0 + 16));
        const f16* bp = Bs + ((kc * 16 + nt) * 32 + L) * 16;
        v16h bf = pack16(*(const v8h*)bp, *(const v8h*)(bp + 8));
        acc = wmma_f16(af, bf, acc);
    }
    int col = nt * 16 + Lm;
    float bv = bias[col];
#pragma unroll
    for (int r = 0; r < 8; ++r) {
        float v = acc[r] + bv;
        if (doRelu) v = fmaxf(v, 0.f);
        out[(mt * 16 + r + 8 * lh) * 256 + col] = (f16)v;
    }
}

// ---------------------------------------------------------------------------
// Fused edge kernel. Block = (b, i). 8 waves, each owns two 16-row j-tiles.
// A (16 pairs x 288 feats) built in registers; B (W3) staged in LDS in four
// 36KB N-quarters via the Tensor Data Mover; epilogue folds +b3, relu, @W4,
// +b4 with a shfl reduction.
__global__ void __launch_bounds__(256)
k_edge(const f16* __restrict__ h16, const f16* __restrict__ at16,
       const float* __restrict__ pos, const f16* __restrict__ W3s,
       const float* __restrict__ b3, const float* __restrict__ W4,
       const float* __restrict__ b4, float* __restrict__ out) {
    __shared__ __align__(16) f16 Bsh[9 * 4 * 32 * 16];   // 36 KB

    int tid = threadIdx.x;
    int L = tid & 31, wv = tid >> 5;
    int Lm = L & 15, lh = L >> 4;
    int b = blockIdx.x >> 8;
    int i = blockIdx.x & 255;

    const f16* hi = h16 + (b * 256 + i) * 256;
    const v8h* Ai = (const v8h*)(at16 + (b * 256 + i) * 16);
    v8h ai0 = Ai[0], ai1 = Ai[1];
    float pix = pos[(b * 256 + i) * 3 + 0];
    float piy = pos[(b * 256 + i) * 3 + 1];
    float piz = pos[(b * 256 + i) * 3 + 2];

    // chunk-0 A-fragments (dist + atom products + zero pad), per j-tile
    v16h afr0[2];
    int jrow[2];
#pragma unroll
    for (int mm = 0; mm < 2; ++mm) {
        int j = (wv * 2 + mm) * 16 + Lm;
        jrow[mm] = j;
        const v8h* Aj = (const v8h*)(at16 + (b * 256 + j) * 16);
        v8h p0 = ai0 * Aj[0];
        v8h p1 = ai1 * Aj[1];
        float dx = pos[(b * 256 + j) * 3 + 0] - pix;
        float dy = pos[(b * 256 + j) * 3 + 1] - piy;
        float dz = pos[(b * 256 + j) * 3 + 2] - piz;
        float d2 = dx * dx + dy * dy + dz * dz;
        f16 d16 = (f16)((d2 > 0.f) ? sqrtf(d2) : 0.f);
        v16h a0 = {};
        if (lh == 0) {           // K = 0..7, 16..23
            a0[0] = d16;
            a0[1] = p0[0]; a0[2] = p0[1]; a0[3] = p0[2]; a0[4] = p0[3];
            a0[5] = p0[4]; a0[6] = p0[5]; a0[7] = p0[6];
            a0[8] = p1[7];       // K=16 -> atom feature 15; K=17..23 -> 0
        } else {                 // K = 8..15, 24..31 (24..31 -> 0)
            a0[0] = p0[7]; a0[1] = p1[0]; a0[2] = p1[1]; a0[3] = p1[2];
            a0[4] = p1[3]; a0[5] = p1[4]; a0[6] = p1[5]; a0[7] = p1[6];
        }
        afr0[mm] = a0;
    }

    float acc2[2][8][4];
#pragma unroll
    for (int mm = 0; mm < 2; ++mm)
#pragma unroll
        for (int r = 0; r < 8; ++r)
#pragma unroll
            for (int e = 0; e < 4; ++e) acc2[mm][r][e] = 0.f;

    for (int nq = 0; nq < 4; ++nq) {
        __syncthreads();
#ifdef USE_TDM
        // One TDM descriptor moves the whole 36KB quarter (9 strided rows of
        // 4096B) into LDS; only wave 0 issues it, TENSORcnt-tracked.
        if (wv == 0) {
            tdm_load_quarter(W3s + (size_t)nq * 2048, Bsh);
            __builtin_amdgcn_s_wait_tensorcnt((short)0);
        }
#else
        // fallback: stage this N-quarter of swizzled W3 via VGPRs
        for (int c = tid; c < 2304; c += 256) {
            int t8   = c & 1;
            int lane = (c >> 1) & 31;
            int ntl  = (c >> 6) & 3;
            int kc   = c >> 8;
            const v8h* src = (const v8h*)(W3s + ((kc * 16 + nq * 4 + ntl) * 32 + lane) * 16 + t8 * 8);
            *(v8h*)(Bsh + ((kc * 4 + ntl) * 32 + lane) * 16 + t8 * 8) = *src;
        }
#endif
        __syncthreads();

#pragma unroll
        for (int mm = 0; mm < 2; ++mm) {
            const f16* hj = h16 + (b * 256 + jrow[mm]) * 256;
            v16h afr[9];
            afr[0] = afr0[mm];
#pragma unroll
            for (int kc = 1; kc < 9; ++kc) {   // latent products, built on the fly
                int c0 = (kc - 1) * 32 + lh * 8;
                v8h r1 = *(const v8h*)(hj + c0)      * *(const v8h*)(hi + c0);
                v8h r2 = *(const v8h*)(hj + c0 + 16) * *(const v8h*)(hi + c0 + 16);
                afr[kc] = pack16(r1, r2);
            }
#pragma unroll
            for (int ntl = 0; ntl < 4; ++ntl) {
                v8f acc = {};
#pragma unroll
                for (int kc = 0; kc < 9; ++kc) {
                    const f16* bp = Bsh + ((kc * 4 + ntl) * 32 + L) * 16;
                    v16h bf = pack16(*(const v8h*)bp, *(const v8h*)(bp + 8));
                    acc = wmma_f16(afr[kc], bf, acc);
                }
                // epilogue: +b3, relu, x W4 row -> 4 edge-type partials
                int col = (nq * 4 + ntl) * 16 + Lm;
                float b3c = b3[col];
                float4 w4 = *(const float4*)(W4 + col * 4);
#pragma unroll
                for (int r = 0; r < 8; ++r) {
                    float v = fmaxf(acc[r] + b3c, 0.f);
                    acc2[mm][r][0] += v * w4.x;
                    acc2[mm][r][1] += v * w4.y;
                    acc2[mm][r][2] += v * w4.z;
                    acc2[mm][r][3] += v * w4.w;
                }
            }
        }
    }

    // reduce the 16 column-lanes of each half-wave, add b4, store
    float4 b4v = *(const float4*)b4;
#pragma unroll
    for (int mm = 0; mm < 2; ++mm) {
#pragma unroll
        for (int r = 0; r < 8; ++r) {
#pragma unroll
            for (int e = 0; e < 4; ++e) {
                float s = acc2[mm][r][e];
                s += __shfl_xor(s, 1);
                s += __shfl_xor(s, 2);
                s += __shfl_xor(s, 4);
                s += __shfl_xor(s, 8);
                acc2[mm][r][e] = s;
            }
        }
        if (Lm == 0) {
            int j0 = (wv * 2 + mm) * 16 + 8 * lh;
#pragma unroll
            for (int r = 0; r < 8; ++r) {
                float4 o;
                o.x = acc2[mm][r][0] + b4v.x;
                o.y = acc2[mm][r][1] + b4v.y;
                o.z = acc2[mm][r][2] + b4v.z;
                o.w = acc2[mm][r][3] + b4v.w;
                *(float4*)(out + (size_t)((b * 256 + i) * 256 + j0 + r) * 4) = o;
            }
        }
    }
}

// ---------------------------------------------------------------------------
extern "C" void kernel_launch(void* const* d_in, const int* in_sizes, int n_in,
                              void* d_out, int out_size, void* d_ws, size_t ws_size,
                              hipStream_t stream) {
    (void)in_sizes; (void)n_in; (void)out_size; (void)ws_size;
    const float* latent    = (const float*)d_in[0];
    const float* positions = (const float*)d_in[1];
    const float* atoms     = (const float*)d_in[2];
    const float* W1 = (const float*)d_in[3];
    const float* b1 = (const float*)d_in[4];
    const float* W2 = (const float*)d_in[5];
    const float* b2 = (const float*)d_in[6];
    const float* W3 = (const float*)d_in[7];
    const float* b3 = (const float*)d_in[8];
    const float* W4 = (const float*)d_in[9];
    const float* b4 = (const float*)d_in[10];
    float* out = (float*)d_out;

    // workspace layout (bytes), ~3.5 MB total, all 16B-aligned
    char* ws = (char*)d_ws;
    f16* lat16 = (f16*)(ws + 0);        // 8*256*256 f16      = 1,048,576 B
    f16* at16  = (f16*)(ws + 1048576);  // 8*256*16 f16       =    65,536 B
    f16* W1s   = (f16*)(ws + 1114112);  // swizzled W1        =   131,072 B
    f16* W2s   = (f16*)(ws + 1245184);  // swizzled W2        =   131,072 B
    f16* W3s   = (f16*)(ws + 1376256);  // swizzled W3 (K288) =   147,456 B
    f16* h1    = (f16*)(ws + 1523712);  // layer-1 activ f16  = 1,048,576 B
    f16* h16   = (f16*)(ws + 2572288);  // h (layer-2) f16    = 1,048,576 B

    k_cvt   <<<2176, 256, 0, stream>>>(latent, atoms, lat16, at16);
    k_swzW12<<< 512, 256, 0, stream>>>(W1, W2, W1s, W2s);
    k_swzW3 <<< 288, 256, 0, stream>>>(W3, W3s);
    k_mlp   <<< 256, 256, 0, stream>>>(lat16, W1s, b1, h1, 1);
    k_mlp   <<< 256, 256, 0, stream>>>(h1,    W2s, b2, h16, 0);
    k_edge  <<<2048, 256, 0, stream>>>(h16, at16, positions, W3s, b3, W4, b4, out);
}
